// GATLayer_21655225106563
// MI455X (gfx1250) — matrix operand order repaired and checked by reading, not compiled, not run on hardware
//
#include <hip/hip_runtime.h>
#include <hip/hip_bf16.h>

typedef __attribute__((ext_vector_type(16))) _Float16 v16h;
typedef __attribute__((ext_vector_type(8)))  float    v8f;

#define F_IN   256
#define HEADS  8
#define CHAN   32       // channels per head
#define FOUT   256      // HEADS*CHAN
#define NEG_SLOPE 0.2f

// ---------- ordered-float <-> uint keys for atomicMax on floats ----------
__device__ __forceinline__ unsigned f2ord(float f) {
  unsigned u = __float_as_uint(f);
  return (u & 0x80000000u) ? ~u : (u | 0x80000000u);
}
__device__ __forceinline__ float ord2f(unsigned u) {
  unsigned v = (u & 0x80000000u) ? (u & 0x7FFFFFFFu) : ~u;
  return __uint_as_float(v);
}
// key of -inf : f2ord(0xFF800000) = 0x007FFFFF
#define NEG_INF_KEY 0x007FFFFFu

// ---------- init kernels ----------
__global__ void gat_zero_f32(float* __restrict__ p, int n) {
  int i = blockIdx.x * blockDim.x + threadIdx.x;
  if (i < n) p[i] = 0.0f;
}
__global__ void gat_init_softmax(unsigned* __restrict__ amax, float* __restrict__ denom, int n) {
  int i = blockIdx.x * blockDim.x + threadIdx.x;
  if (i < n) { amax[i] = NEG_INF_KEY; denom[i] = 0.0f; }
}

// ---------- pack W (f32 [K=256][N=256]) into per-lane WMMA B fragments ----------
// Fragment layout (16-bit B, 32x16 per chunk), this code's contiguous k order, same as the A fragment:
//   lane = kg*16 + n :  lanes 0-15 hold K = kc*32 + 0..15, lanes 16-31 hold K = kc*32 + 16..31
//   elem e in v16h   :  K_local = kg*16 + e
// Wpack flat index: (((kc*16 + ct)*32 + lane)*16 + e), kc=K-chunk(8), ct=col-tile(16)
__global__ void gat_pack_W(const float* __restrict__ W, _Float16* __restrict__ Wpack) {
  int i = blockIdx.x * blockDim.x + threadIdx.x;     // 0 .. 65535
  if (i >= 65536) return;
  int e    = i & 15;
  int lane = (i >> 4) & 31;
  int ct   = (i >> 9) & 15;
  int kc   = i >> 13;
  int n  = lane & 15;
  int kg = lane >> 4;
  int k   = kc * 32 + kg * 16 + e;
  int col = ct * 16 + n;
  Wpack[i] = (_Float16)W[k * FOUT + col];
}

// ---------- GEMM: h16[N][256] = f16( x[N][256] @ W[256][256] ) ----------
// Block = 128 threads = 4 waves. Block b owns rows [16b, 16b+16).
// Wave w owns cols [64w, 64w+64): 1 A-fragment reused against 4 B-fragments.
__global__ __launch_bounds__(128)
void gat_gemm_wmma(const float* __restrict__ x, const _Float16* __restrict__ Wpack,
                   _Float16* __restrict__ h16, int N) {
  const int lane = threadIdx.x & 31;
  const int wv   = threadIdx.x >> 5;            // 0..3
  const int row0 = blockIdx.x * 16;
  const int m    = lane & 15;
  const int kg   = lane >> 4;

  int arow = row0 + m;
  if (arow >= N) arow = N - 1;                  // clamp (EXEC must stay all-1 for WMMA)
  const float* __restrict__ xrow = x + (size_t)arow * F_IN;

  v8f acc0 = {}, acc1 = {}, acc2 = {}, acc3 = {};

#pragma unroll
  for (int kc = 0; kc < 8; ++kc) {
    // A fragment: lane holds row m, K = kc*32 + kg*16 + 0..15
    const float* p = xrow + kc * 32 + kg * 16;
    float4 f0 = ((const float4*)p)[0];
    float4 f1 = ((const float4*)p)[1];
    float4 f2 = ((const float4*)p)[2];
    float4 f3 = ((const float4*)p)[3];
    v16h a;
    a[0]  = (_Float16)f0.x; a[1]  = (_Float16)f0.y; a[2]  = (_Float16)f0.z; a[3]  = (_Float16)f0.w;
    a[4]  = (_Float16)f1.x; a[5]  = (_Float16)f1.y; a[6]  = (_Float16)f1.z; a[7]  = (_Float16)f1.w;
    a[8]  = (_Float16)f2.x; a[9]  = (_Float16)f2.y; a[10] = (_Float16)f2.z; a[11] = (_Float16)f2.w;
    a[12] = (_Float16)f3.x; a[13] = (_Float16)f3.y; a[14] = (_Float16)f3.z; a[15] = (_Float16)f3.w;

    const _Float16* bbase = Wpack + (((size_t)kc * 16 + wv * 4) * 32 + lane) * 16;
    v16h b0 = *(const v16h*)(bbase);
    v16h b1 = *(const v16h*)(bbase + 1 * 32 * 16);
    v16h b2 = *(const v16h*)(bbase + 2 * 32 * 16);
    v16h b3 = *(const v16h*)(bbase + 3 * 32 * 16);

    acc0 = __builtin_amdgcn_wmma_f32_16x16x32_f16(false, a, false, b0, (short)0, acc0, false, false);
    acc1 = __builtin_amdgcn_wmma_f32_16x16x32_f16(false, a, false, b1, (short)0, acc1, false, false);
    acc2 = __builtin_amdgcn_wmma_f32_16x16x32_f16(false, a, false, b2, (short)0, acc2, false, false);
    acc3 = __builtin_amdgcn_wmma_f32_16x16x32_f16(false, a, false, b3, (short)0, acc3, false, false);
  }

  // D layout: VGPR r -> row (row0 + 8*kg + r), col = wv*64 + s*16 + (lane&15)
  const int n = lane & 15;
#pragma unroll
  for (int r = 0; r < 8; ++r) {
    int orow = row0 + 8 * kg + r;
    if (orow < N) {
      _Float16* o = h16 + (size_t)orow * FOUT + wv * 64 + n;
      o[0]  = (_Float16)acc0[r];
      o[16] = (_Float16)acc1[r];
      o[32] = (_Float16)acc2[r];
      o[48] = (_Float16)acc3[r];
    }
  }
}

// ---------- per-node attention logits: a_src[n,h], a_dst[n,h] ----------
__global__ void gat_attn_coef(const _Float16* __restrict__ h16,
                              const float* __restrict__ att_src,
                              const float* __restrict__ att_dst,
                              float* __restrict__ a_src, float* __restrict__ a_dst, int NH) {
  int i = blockIdx.x * blockDim.x + threadIdx.x;
  if (i >= NH) return;
  int head = i & (HEADS - 1);
  int node = i >> 3;
  const _Float16* hp = h16 + (size_t)node * FOUT + head * CHAN;
  const float* as = att_src + head * CHAN;
  const float* ad = att_dst + head * CHAN;
  float s1 = 0.f, s2 = 0.f;
#pragma unroll
  for (int c = 0; c < CHAN; ++c) {
    float v = (float)hp[c];
    s1 += v * as[c];
    s2 += v * ad[c];
  }
  a_src[i] = s1;
  a_dst[i] = s2;
}

__device__ __forceinline__ void edge_endpoints(const int* __restrict__ ei, int e, int E,
                                               int& s, int& d) {
  if (e < E) { s = ei[e]; d = ei[E + e]; }
  else       { s = e - E; d = e - E; }          // self loops appended
}

// ---------- pass 1: per-(dst,head) running max of leaky_relu(alpha) ----------
__global__ void gat_edge_max(const int* __restrict__ ei,
                             const float* __restrict__ a_src, const float* __restrict__ a_dst,
                             unsigned* __restrict__ amax, int E, int E2H) {
  int i = blockIdx.x * blockDim.x + threadIdx.x;
  if (i >= E2H) return;
  int head = i & (HEADS - 1);
  int e = i >> 3;
  int s, d; edge_endpoints(ei, e, E, s, d);
  float al = a_src[s * HEADS + head] + a_dst[d * HEADS + head];
  al = (al > 0.f) ? al : NEG_SLOPE * al;
  atomicMax(&amax[d * HEADS + head], f2ord(al));
}

// ---------- pass 2: ex = exp(alpha - max), denom += ex, stash ex ----------
__global__ void gat_edge_exp(const int* __restrict__ ei,
                             const float* __restrict__ a_src, const float* __restrict__ a_dst,
                             const unsigned* __restrict__ amax, float* __restrict__ denom,
                             float* __restrict__ exbuf, int E, int E2H) {
  int i = blockIdx.x * blockDim.x + threadIdx.x;
  if (i >= E2H) return;
  int head = i & (HEADS - 1);
  int e = i >> 3;
  int s, d; edge_endpoints(ei, e, E, s, d);
  float al = a_src[s * HEADS + head] + a_dst[d * HEADS + head];
  al = (al > 0.f) ? al : NEG_SLOPE * al;
  float mx = ord2f(amax[d * HEADS + head]);
  float ex = __expf(al - mx);
  exbuf[i] = ex;
  atomicAdd(&denom[d * HEADS + head], ex);
}

// ---------- pass 3: out[dst] += h[src] * attn  (1 block of 256 threads per edge) ----------
__global__ __launch_bounds__(256)
void gat_scatter(const int* __restrict__ ei, const _Float16* __restrict__ h16,
                 const float* __restrict__ exbuf, const float* __restrict__ denom,
                 float* __restrict__ out, int E) {
  int e = blockIdx.x;
  int t = threadIdx.x;                   // 0..255 -> head = t>>5, chan = t&31
  int head = t >> 5;
  int s, d; edge_endpoints(ei, e, E, s, d);
  float attn = exbuf[(size_t)e * HEADS + head] /
               (denom[d * HEADS + head] + 1e-16f);
  float v = (float)h16[(size_t)s * FOUT + t] * attn;
  atomicAdd(&out[(size_t)d * FOUT + t], v);
}

// ---------- finalize: out = elu(out + bias) ----------
__global__ void gat_finalize(float* __restrict__ out, const float* __restrict__ bias, int n) {
  int i = blockIdx.x * blockDim.x + threadIdx.x;
  if (i >= n) return;
  float o = out[i] + bias[i & (FOUT - 1)];
  out[i] = (o > 0.f) ? o : (__expf(o) - 1.f);
}

extern "C" void kernel_launch(void* const* d_in, const int* in_sizes, int n_in,
                              void* d_out, int out_size, void* d_ws, size_t ws_size,
                              hipStream_t stream) {
  const float* x       = (const float*)d_in[0];
  const int*   ei      = (const int*)  d_in[1];
  const float* W       = (const float*)d_in[2];
  const float* att_src = (const float*)d_in[3];
  const float* att_dst = (const float*)d_in[4];
  const float* bias    = (const float*)d_in[5];
  float* out = (float*)d_out;

  const int N   = in_sizes[0] / F_IN;     // 50000
  const int E   = in_sizes[1] / 2;        // 800000
  const int E2  = E + N;                  // + self loops
  const int NH  = N * HEADS;
  const int E2H = E2 * HEADS;
  const int NO  = N * FOUT;

  // ---- carve workspace (256B aligned) ----
  char* wp = (char*)d_ws;
  auto carve = [&](size_t bytes) -> void* {
    void* p = (void*)wp;
    wp += (bytes + 255) & ~(size_t)255;
    return p;
  };
  _Float16* Wpack = (_Float16*)carve(65536 * sizeof(_Float16));
  _Float16* h16   = (_Float16*)carve((size_t)N * FOUT * sizeof(_Float16));
  float*    a_src = (float*)   carve((size_t)NH * sizeof(float));
  float*    a_dst = (float*)   carve((size_t)NH * sizeof(float));
  unsigned* amax  = (unsigned*)carve((size_t)NH * sizeof(unsigned));
  float*    denom = (float*)   carve((size_t)NH * sizeof(float));
  float*    exbuf = (float*)   carve((size_t)E2H * sizeof(float));
  (void)ws_size; (void)n_in; (void)out_size;

  const int T = 256;

  // init accumulators (d_out is poisoned / left dirty between replays)
  gat_zero_f32<<<(NO + T - 1) / T, T, 0, stream>>>(out, NO);
  gat_init_softmax<<<(NH + T - 1) / T, T, 0, stream>>>(amax, denom, NH);

  // GEMM path
  gat_pack_W<<<65536 / T, T, 0, stream>>>(W, Wpack);
  gat_gemm_wmma<<<(N + 15) / 16, 128, 0, stream>>>(x, Wpack, h16, N);

  // attention logits
  gat_attn_coef<<<(NH + T - 1) / T, T, 0, stream>>>(h16, att_src, att_dst, a_src, a_dst, NH);

  // softmax over incoming edges
  gat_edge_max<<<(E2H + T - 1) / T, T, 0, stream>>>(ei, a_src, a_dst, amax, E, E2H);
  gat_edge_exp<<<(E2H + T - 1) / T, T, 0, stream>>>(ei, a_src, a_dst, amax, denom, exbuf, E, E2H);

  // weighted message scatter
  gat_scatter<<<E2, 256, 0, stream>>>(ei, h16, exbuf, denom, out, E);

  // bias + ELU
  gat_finalize<<<(NO + T - 1) / T, T, 0, stream>>>(out, bias, NO);
}
